// PointNextPatchTokenizer_19224273617479
// MI455X (gfx1250) — matrix operand order repaired and checked by reading, not compiled, not run on hardware
//
#include <hip/hip_runtime.h>
#include <math.h>

#define DEV __device__ __forceinline__

typedef __attribute__((ext_vector_type(16))) _Float16 v16h;
typedef __attribute__((ext_vector_type(8)))  _Float16 v8h;
typedef __attribute__((ext_vector_type(2)))  _Float16 v2h;
typedef __attribute__((ext_vector_type(8)))  float    v8f;

constexpr int cB = 32, cN = 16384, cS = 64;

DEV float gelu_f(float x){
  // jax.nn.gelu default (tanh approximation)
  return 0.5f*x*(1.f + tanhf(0.7978845608028654f*(x + 0.044715f*x*x*x)));
}

DEV v8f wmma_f16(v16h a, v16h b, v8f c){
  // v_wmma_f32_16x16x32_f16  D = A(16x32 f16) * B(32x16 f16) + C(16x16 f32)
  return __builtin_amdgcn_wmma_f32_16x16x32_f16(false, a, false, b, (short)0, c, false, false);
}

DEV v16h concat8(v8h lo, v8h hi){
  return __builtin_shufflevector(lo, hi, 0,1,2,3,4,5,6,7,8,9,10,11,12,13,14,15);
}

// ---- CDNA5 async global->LDS copy (ASYNCcnt path, ISA 08_async_tensor §4) ----
// LDS dest address: low 32 bits of the generic pointer ARE the workgroup-relative
// LDS byte offset.
DEV void async_load_b128(const void* gsrc, void* ldst){
  unsigned loff = (unsigned)(unsigned long long)ldst;
  asm volatile("global_load_async_to_lds_b128 %0, %1, off"
               :: "v"(loff), "v"(gsrc)
               : "memory");
}
DEV void async_wait0(){
  asm volatile("s_wait_asynccnt 0" ::: "memory");
}

// A fragment: 16x32 f16 tile from row-major matrix (ISA 7.12.2 layout).
// lane (g,m): elements 0..7  = K 8g..8g+7   (one b128 load)
//             elements 8..15 = K 16+8g..+7  (one b128 load)
DEV v16h load_Afrag(const _Float16* A, int lda, int k0, int lane){
  int g = lane >> 4, m = lane & 15;
  const _Float16* base = A + m*lda + k0 + 8*g;
  v8h lo = *(const v8h*)(base);
  v8h hi = *(const v8h*)(base + 16);
  return concat8(lo, hi);
}

// B fragment from TRANSPOSED weights WT[N][Kpad] (row-major in N).
// lane (g,n): elements 0..7 = K k0+8g..+7, elements 8..15 = K k0+16+8g..+7 (two b128 loads)
DEV v16h load_BfragT(const _Float16* WT, int ldk, int k0, int n0, int lane){
  int g = lane >> 4, n = lane & 15;
  const _Float16* base = WT + (n0 + n)*ldk + k0 + 8*g;
  v8h lo = *(const v8h*)(base);
  v8h hi = *(const v8h*)(base + 16);
  return concat8(lo, hi);
}

// -------- weight prep: f32 [K][N] -> f16 transposed [N][Kpad] (zero pad K) ----
__global__ void convert_T_f16(const float* __restrict__ src, _Float16* __restrict__ dst,
                              int srowsK, int cols, int Kpad){
  int i = blockIdx.x*blockDim.x + threadIdx.x;
  if (i >= cols*Kpad) return;
  int n = i / Kpad, k = i % Kpad;
  dst[i] = (k < srowsK) ? (_Float16)src[k*cols + n] : (_Float16)0.f;
}

// ---------------- stem: pmlp(6->64,gelu) ; pmlp(64->64,gelu) via WMMA ----------
__global__ __launch_bounds__(256) void stem_kernel(
    const float* __restrict__ pc,
    const float* __restrict__ W1, const float* __restrict__ b1,
    const float* __restrict__ g1, const float* __restrict__ be1,
    const float* __restrict__ b2, const float* __restrict__ g2,
    const float* __restrict__ be2,
    const _Float16* __restrict__ W2T,   // transposed [64][64] f16
    _Float16* __restrict__ feat)
{
  __shared__ float sW1[6*64];
  __shared__ float svec[6*64];                       // b1,g1,be1,b2,g2,be2
  __shared__ __align__(16) _Float16 sW2[64*64];      // transposed [N][K]
  __shared__ __align__(16) _Float16 sA[8][16*64];    // per-wave tile

  int tid = threadIdx.x;
  // async global->LDS stage of the transposed stem2 weights (8KB)
  for (int i = tid; i < 512; i += 256)
    async_load_b128(W2T + i*8, &sW2[i*8]);
  for (int i = tid; i < 6*64; i += 256) sW1[i] = W1[i];
  if (tid < 64){
    svec[tid]     = b1[tid];  svec[64+tid]  = g1[tid]; svec[128+tid] = be1[tid];
    svec[192+tid] = b2[tid];  svec[256+tid] = g2[tid]; svec[320+tid] = be2[tid];
  }
  async_wait0();
  __syncthreads();

  int w = tid >> 5, lane = tid & 31;
  {
    int g = lane >> 4, m = lane & 15;
    long pt = (long)blockIdx.x*128 + w*16 + m;
    float x[6];
#pragma unroll
    for (int i=0;i<6;i++) x[i] = pc[pt*6 + i];
    float y[32];
    float s1 = 0.f, s2 = 0.f;
#pragma unroll
    for (int c=0;c<32;c++){
      int ch = g*32 + c;
      float acc = svec[ch];
#pragma unroll
      for (int i=0;i<6;i++) acc += x[i]*sW1[i*64 + ch];
      y[c] = acc; s1 += acc; s2 += acc*acc;
    }
    s1 += __shfl_xor(s1, 16);
    s2 += __shfl_xor(s2, 16);
    float mu = s1*(1.f/64.f);
    float var = s2*(1.f/64.f) - mu*mu;
    float rs = rsqrtf(var + 1e-5f);
    v8h* dst = (v8h*)&sA[w][m*64 + g*32];
#pragma unroll
    for (int q=0;q<4;q++){
      v8h t;
#pragma unroll
      for (int e=0;e<8;e++){
        int ch = g*32 + q*8 + e;
        float v = (y[q*8+e]-mu)*rs*svec[64+ch] + svec[128+ch];
        t[e] = (_Float16)gelu_f(v);
      }
      dst[q] = t;
    }
  }
  __syncthreads();

  v16h a0 = load_Afrag(sA[w], 64, 0,  lane);
  v16h a1 = load_Afrag(sA[w], 64, 32, lane);
  v8f cts[4];
  float sr1[8], sr2[8];
#pragma unroll
  for (int r=0;r<8;r++){ sr1[r]=0.f; sr2[r]=0.f; }
  int n = lane & 15, gg = lane >> 4;
#pragma unroll
  for (int t=0;t<4;t++){
    v16h bf0 = load_BfragT(sW2, 64, 0,  t*16, lane);
    v16h bf1 = load_BfragT(sW2, 64, 32, t*16, lane);
    v8f c = {};
    c = wmma_f16(a0, bf0, c);
    c = wmma_f16(a1, bf1, c);
    float bias = svec[192 + t*16 + n];
#pragma unroll
    for (int r=0;r<8;r++){ float v = c[r] + bias; c[r]=v; sr1[r]+=v; sr2[r]+=v*v; }
    cts[t] = c;
  }
#pragma unroll
  for (int d=1; d<16; d<<=1){
#pragma unroll
    for (int r=0;r<8;r++){ sr1[r]+=__shfl_xor(sr1[r], d); sr2[r]+=__shfl_xor(sr2[r], d); }
  }
  // LN + gelu -> restage into sA (f16), then wide coalesced global stores
#pragma unroll
  for (int r=0;r<8;r++){
    float mu = sr1[r]*(1.f/64.f);
    float var = sr2[r]*(1.f/64.f) - mu*mu;
    float rs = rsqrtf(var + 1e-5f);
#pragma unroll
    for (int t=0;t<4;t++){
      int ch = t*16 + n;
      float v = (cts[t][r]-mu)*rs*svec[256+ch] + svec[320+ch];
      sA[w][(r + 8*gg)*64 + ch] = (_Float16)gelu_f(v);
    }
  }
  __syncthreads();
  long gbase = ((long)blockIdx.x*128 + w*16) * 64;
#pragma unroll
  for (int q=0;q<4;q++){
    v8h t = *(const v8h*)&sA[w][lane*32 + q*8];
    *(v8h*)(feat + gbase + lane*32 + q*8) = t;
  }
}

// ---------------- farthest point sampling: 1 block per batch -------------------
__global__ __launch_bounds__(1024) void fps_kernel(const float* __restrict__ pc,
    int* __restrict__ cidx, float* __restrict__ pcen)
{
  int b = blockIdx.x, tid = threadIdx.x;
  const float* xyz = pc + (long)b*cN*6;
  float px[16], py[16], pz[16], dist[16];
#pragma unroll
  for (int j=0;j<16;j++){
    int p = tid + j*1024;
    px[j]=xyz[p*6+0]; py[j]=xyz[p*6+1]; pz[j]=xyz[p*6+2];
    dist[j]=1e10f;
  }
  __shared__ float smax[1024];
  __shared__ int   simx[1024];
  __shared__ float cen[3];
  __shared__ int   sfar;
  if (tid==0) sfar = 0;
  __syncthreads();
  for (int s=0; s<cS; s++){
    int far = sfar;
    if (tid==0){
      cidx[b*cS+s] = far;
      float cx=xyz[far*6], cy=xyz[far*6+1], cz=xyz[far*6+2];
      cen[0]=cx; cen[1]=cy; cen[2]=cz;
      pcen[((long)b*cS+s)*3+0]=cx; pcen[((long)b*cS+s)*3+1]=cy; pcen[((long)b*cS+s)*3+2]=cz;
    }
    __syncthreads();
    float cx=cen[0], cy=cen[1], cz=cen[2];
    float best=-1.f; int bi=0;
#pragma unroll
    for (int j=0;j<16;j++){
      float dx=px[j]-cx, dy=py[j]-cy, dz=pz[j]-cz;
      float d = dx*dx+dy*dy+dz*dz;
      dist[j] = fminf(dist[j], d);
      if (dist[j] > best){ best = dist[j]; bi = tid + j*1024; }
    }
    smax[tid]=best; simx[tid]=bi;
    __syncthreads();
    for (int st=512; st>0; st>>=1){
      if (tid < st){
        float ov=smax[tid+st]; int oi=simx[tid+st];
        if (ov>smax[tid] || (ov==smax[tid] && oi<simx[tid])){ smax[tid]=ov; simx[tid]=oi; }
      }
      __syncthreads();
    }
    if (tid==0) sfar = simx[0];
    __syncthreads();
  }
}

// ---------------- ball query: one wave per center, ordered compaction ----------
__global__ __launch_bounds__(256) void ball_kernel(const float* __restrict__ pc,
    const float* __restrict__ pcen, int* __restrict__ nidx)
{
  int w = threadIdx.x >> 5, lane = threadIdx.x & 31;
  int center = blockIdx.x*8 + w;
  int b = center >> 6;
  const float* xyz = pc + (long)b*cN*6;
  float cx=pcen[center*3+0], cy=pcen[center*3+1], cz=pcen[center*3+2];
  int* out0 = nidx + (long)center*48;
  int* out1 = out0 + 16;
  const float r0sq = 0.04f*0.04f, r1sq = 0.08f*0.08f;
  int cnt0=0, cnt1=0;
  unsigned below = (lane==0) ? 0u : ((1u<<lane)-1u);
  for (int base=0; base<cN && (cnt0<16 || cnt1<32); base+=32){
    int p = base + lane;
    float dx=xyz[p*6+0]-cx, dy=xyz[p*6+1]-cy, dz=xyz[p*6+2]-cz;
    float d2 = dx*dx+dy*dy+dz*dz;
    unsigned mm0 = (unsigned)__ballot(d2 <= r0sq);
    unsigned mm1 = (unsigned)__ballot(d2 <= r1sq);
    if (cnt0 < 16 && mm0){
      int pos = cnt0 + __popc(mm0 & below);
      if (((mm0>>lane)&1u) && pos < 16) out0[pos] = p;
      cnt0 = min(16, cnt0 + __popc(mm0));
    }
    if (cnt1 < 32 && mm1){
      int pos = cnt1 + __popc(mm1 & below);
      if (((mm1>>lane)&1u) && pos < 32) out1[pos] = p;
      cnt1 = min(32, cnt1 + __popc(mm1));
    }
  }
  int first0 = (cnt0 > 0) ? out0[0] : (cN-1);
  int first1 = (cnt1 > 0) ? out1[0] : (cN-1);
  for (int i = cnt0 + lane; i < 16; i += 32) out0[i] = first0;
  for (int i = cnt1 + lane; i < 32; i += 32) out1[i] = first1;
}

// ---------------- grouping MLPs (both scales), block per center ----------------
struct GroupArgs {
  const float* pc; const _Float16* feat; const float* pcen; const int* nidx;
  const _Float16 *w0l1,*w0l2,*w1l1,*w1l2;   // transposed [N][Kpad]
  const float *b0l1,*g0l1,*be0l1;
  const float *b0l2,*g0l2,*be0l2;
  const float *b1l1,*g1l1,*be1l1;
  const float *b1l2,*g1l2,*be1l2;
  float* sfeat;
};

__global__ __launch_bounds__(256) void group_kernel(GroupArgs A){
  __shared__ __align__(16) _Float16 gin[48*96];  // rows 0-15 scale0, 16-47 scale1
  __shared__ __align__(16) _Float16 h1s[48*128];
  __shared__ float yb[48*128];
  __shared__ float smu[48], srs[48];

  int tid = threadIdx.x, w = tid>>5, lane = tid&31;
  int center = blockIdx.x;
  int b = center >> 6;
  const float* xyz = A.pc + (long)b*cN*6;
  const int* ni = A.nidx + (long)center*48;
  float cx=A.pcen[center*3+0], cy=A.pcen[center*3+1], cz=A.pcen[center*3+2];
  if (tid == 0){
    __builtin_prefetch(A.w0l1, 0, 0); __builtin_prefetch(A.w0l2, 0, 0);
    __builtin_prefetch(A.w1l1, 0, 0); __builtin_prefetch(A.w1l2, 0, 0);
  }

  for (int row = w; row < 48; row += 8){
    int p = ni[row];
    const _Float16* fp = A.feat + ((long)b*cN + p)*64;
    // async gather of the 128B neighbor feature row into LDS
    if (lane < 8)
      async_load_b128(fp + lane*8, &gin[row*96 + lane*8]);
    if (lane >= 8 && lane < 11){
      int cl = lane - 8;
      float ccv = (cl==0)?cx:((cl==1)?cy:cz);
      float rel = xyz[p*6 + cl] - ccv;
      gin[row*96 + 64 + cl] = (_Float16)rel;
      const float fr[4] = {1.f, 0.1f, 0.01f, 0.001f};  // 10000^(-t/4)
#pragma unroll
      for (int t=0;t<4;t++){
        float ang = rel*fr[t];
        gin[row*96 + 67 + cl*8 + t]     = (_Float16)sinf(ang);
        gin[row*96 + 67 + cl*8 + 4 + t] = (_Float16)cosf(ang);
      }
    }
    if (lane >= 27) gin[row*96 + 64 + lane] = (_Float16)0.f;  // pad cols 91..95
  }
  async_wait0();
  __syncthreads();

  for (int pass = 0; pass < 2; ++pass){
    for (int job = w; job < 24; job += 8){
      int sc = (job < 8) ? 0 : 1;
      int m0 = (job < 8) ? 0 : ((job < 16) ? 16 : 32);
      int nt = (job < 8) ? job : ((job < 16) ? job - 8 : job - 16);
      v8f c = {};
      if (pass == 0){
        const _Float16* Wt = sc ? A.w1l1 : A.w0l1;
#pragma unroll
        for (int ks = 0; ks < 3; ++ks){
          v16h af = load_Afrag(gin + m0*96, 96, ks*32, lane);
          v16h bf = load_BfragT(Wt, 96, ks*32, nt*16, lane);
          c = wmma_f16(af, bf, c);
        }
      } else {
        const _Float16* Wt = sc ? A.w1l2 : A.w0l2;
#pragma unroll
        for (int ks = 0; ks < 4; ++ks){
          v16h af = load_Afrag(h1s + m0*128, 128, ks*32, lane);
          v16h bf = load_BfragT(Wt, 128, ks*32, nt*16, lane);
          c = wmma_f16(af, bf, c);
        }
      }
      const float* bb = (pass==0) ? (sc ? A.b1l1 : A.b0l1) : (sc ? A.b1l2 : A.b0l2);
      int n = lane & 15, gg = lane >> 4;
      float bias = bb[nt*16 + n];
#pragma unroll
      for (int r = 0; r < 8; ++r)
        yb[(m0 + r + 8*gg)*128 + nt*16 + n] = c[r] + bias;
    }
    __syncthreads();
    if (tid < 48){
      float s1=0.f, s2=0.f;
      for (int cc=0; cc<128; ++cc){ float v = yb[tid*128+cc]; s1+=v; s2+=v*v; }
      float mu = s1*(1.f/128.f);
      float var = s2*(1.f/128.f) - mu*mu;
      smu[tid]=mu; srs[tid]=rsqrtf(var + 1e-5f);
    }
    __syncthreads();
    if (pass == 0){
      for (int i = tid*2; i < 48*128; i += 512){
        int row = i >> 7, cc = i & 127;
        const float* gv  = (row < 16) ? A.g0l1  : A.g1l1;
        const float* bev = (row < 16) ? A.be0l1 : A.be1l1;
        float v0 = (yb[i]  -smu[row])*srs[row]*gv[cc]   + bev[cc];
        float v1 = (yb[i+1]-smu[row])*srs[row]*gv[cc+1] + bev[cc+1];
        v2h t; t[0] = (_Float16)gelu_f(v0); t[1] = (_Float16)gelu_f(v1);
        *(v2h*)&h1s[i] = t;
      }
      __syncthreads();
    }
  }
  // l2 has act=False: LN only, then max over neighbors
  if (tid < 128){
    int cc = tid;
    float mx = -3.4e38f;
    for (int r = 0; r < 16; ++r){
      float v = (yb[r*128+cc]-smu[r])*srs[r]*A.g0l2[cc] + A.be0l2[cc];
      mx = fmaxf(mx, v);
    }
    A.sfeat[(long)center*256 + cc] = mx;
  } else {
    int cc = tid - 128;
    float mx = -3.4e38f;
    for (int r = 16; r < 48; ++r){
      float v = (yb[r*128+cc]-smu[r])*srs[r]*A.g1l2[cc] + A.be1l2[cc];
      mx = fmaxf(mx, v);
    }
    A.sfeat[(long)center*256 + 128 + cc] = mx;
  }
}

// ---------------- final projection: cat(416) -> 128 -> 128 ---------------------
struct FinalArgs {
  const _Float16* feat; const int* cidx; const float* pcen; const float* sfeat;
  const _Float16 *wp1,*wp2;   // transposed [N][Kpad]
  const float *bp1,*gp1,*bep1, *bp2,*gp2,*bep2;
  float* out;
};

__global__ __launch_bounds__(256) void final_kernel(FinalArgs A){
  __shared__ __align__(16) _Float16 cat[16*416];
  __shared__ __align__(16) _Float16 h1s[16*128];
  __shared__ float yb[16*128];
  __shared__ float smu[16], srs[16];

  int tid = threadIdx.x, w = tid>>5, lane = tid&31;
  int cbase = blockIdx.x * 16;
  if (tid == 0){
    __builtin_prefetch(A.wp1, 0, 0); __builtin_prefetch(A.wp2, 0, 0);
  }

  for (int row = w; row < 16; row += 8){
    int center = cbase + row;
    int b = center >> 6;
    int p = A.cidx[center];
    const _Float16* fp = A.feat + ((long)b*cN + p)*64;
    // async gather of the center feature row into LDS
    if (lane < 8)
      async_load_b128(fp + lane*8, &cat[row*416 + lane*8]);
    {
      v8h t;
#pragma unroll
      for (int e=0;e<8;e++)
        t[e] = (_Float16)A.sfeat[(long)center*256 + lane*8 + e];
      *(v8h*)&cat[row*416 + 64 + lane*8] = t;
    }
    if (lane >= 8 && lane < 11){
      int cl = lane - 8;
      float x = A.pcen[center*3 + cl];
#pragma unroll
      for (int t=0;t<16;t++){
        float f = expf(-9.210340371976184f * (float)t * (1.f/16.f)); // 10000^(-t/16)
        float ang = x*f;
        cat[row*416 + 320 + cl*32 + t]      = (_Float16)sinf(ang);
        cat[row*416 + 320 + cl*32 + 16 + t] = (_Float16)cosf(ang);
      }
    }
  }
  async_wait0();
  __syncthreads();

  int n = lane & 15, gg = lane >> 4;
  for (int pass = 0; pass < 2; ++pass){
    {
      int nt = w;
      v8f c = {};
      if (pass == 0){
#pragma unroll
        for (int ks = 0; ks < 13; ++ks){  // 416 = 13*32 exactly
          v16h af = load_Afrag(cat, 416, ks*32, lane);
          v16h bf = load_BfragT(A.wp1, 416, ks*32, nt*16, lane);
          c = wmma_f16(af, bf, c);
        }
      } else {
#pragma unroll
        for (int ks = 0; ks < 4; ++ks){
          v16h af = load_Afrag(h1s, 128, ks*32, lane);
          v16h bf = load_BfragT(A.wp2, 128, ks*32, nt*16, lane);
          c = wmma_f16(af, bf, c);
        }
      }
      const float* bb = (pass==0) ? A.bp1 : A.bp2;
      float bias = bb[nt*16 + n];
#pragma unroll
      for (int r=0;r<8;r++) yb[(r + 8*gg)*128 + nt*16 + n] = c[r] + bias;
    }
    __syncthreads();
    if (tid < 16){
      float s1=0.f, s2=0.f;
      for (int cc=0; cc<128; ++cc){ float v = yb[tid*128+cc]; s1+=v; s2+=v*v; }
      float mu = s1*(1.f/128.f), var = s2*(1.f/128.f)-mu*mu;
      smu[tid]=mu; srs[tid]=rsqrtf(var+1e-5f);
    }
    __syncthreads();
    if (pass == 0){
      for (int i = tid*2; i < 16*128; i += 512){
        int row = i >> 7, cc = i & 127;
        float v0 = (yb[i]  -smu[row])*srs[row]*A.gp1[cc]   + A.bep1[cc];
        float v1 = (yb[i+1]-smu[row])*srs[row]*A.gp1[cc+1] + A.bep1[cc+1];
        v2h t; t[0] = (_Float16)gelu_f(v0); t[1] = (_Float16)gelu_f(v1);
        *(v2h*)&h1s[i] = t;
      }
      __syncthreads();
    } else {
      for (int i = tid*2; i < 16*128; i += 512){
        int row = i >> 7, cc = i & 127;
        float2 t;
        t.x = (yb[i]  -smu[row])*srs[row]*A.gp2[cc]   + A.bep2[cc];
        t.y = (yb[i+1]-smu[row])*srs[row]*A.gp2[cc+1] + A.bep2[cc+1];
        *(float2*)&A.out[(long)(cbase+row)*128 + cc] = t;
      }
    }
  }
  if (tid < 48)
    A.out[(long)cB*cS*128 + (long)cbase*3 + tid] = A.pcen[(long)cbase*3 + tid];
}

// ---------------- host launch --------------------------------------------------
extern "C" void kernel_launch(void* const* d_in, const int* in_sizes, int n_in,
                              void* d_out, int out_size, void* d_ws, size_t ws_size,
                              hipStream_t stream)
{
  (void)n_in; (void)out_size; (void)ws_size;
  // Detect flattening order: Python insertion order (pointcloud first) vs
  // jax pytree sorted order ('params' < 'pointcloud'; leaves W,b,beta,g).
  bool ins = (in_sizes[0] == cB*cN*6);

  struct PPk { const float *W,*b,*g,*be; };
  auto mk = [&](int base)->PPk{
    PPk p;
    p.W = (const float*)d_in[base];
    p.b = (const float*)d_in[base+1];
    if (ins){ p.g  = (const float*)d_in[base+2]; p.be = (const float*)d_in[base+3]; }
    else    { p.be = (const float*)d_in[base+2]; p.g  = (const float*)d_in[base+3]; }
    return p;
  };
  const float* pc;
  PPk stem1, stem2, s0l1, s0l2, s1l1, s1l2, pr1, pr2;
  if (ins){
    pc = (const float*)d_in[0];
    stem1 = mk(1);  stem2 = mk(5);
    s0l1  = mk(9);  s0l2  = mk(13);
    s1l1  = mk(17); s1l2  = mk(21);
    pr1   = mk(25); pr2   = mk(29);
  } else {
    pr1   = mk(0);  pr2   = mk(4);
    s0l1  = mk(8);  s0l2  = mk(12);
    s1l1  = mk(16); s1l2  = mk(20);
    stem1 = mk(24); stem2 = mk(28);
    pc = (const float*)d_in[32];
  }

  char* ws = (char*)d_ws;
  size_t off = 0;
  _Float16* feat  = (_Float16*)(ws + off); off += (size_t)cB*cN*64*2;
  int*      cidx  = (int*)(ws + off);      off += (size_t)cB*cS*4;
  float*    pcen  = (float*)(ws + off);    off += (size_t)cB*cS*3*4;
  int*      nidx  = (int*)(ws + off);      off += (size_t)cB*cS*48*4;
  float*    sfeat = (float*)(ws + off);    off += (size_t)cB*cS*256*4;
  _Float16* w_stem2 = (_Float16*)(ws + off); off += 64*64*2;
  _Float16* w_s0l1  = (_Float16*)(ws + off); off += 128*96*2;
  _Float16* w_s0l2  = (_Float16*)(ws + off); off += 128*128*2;
  _Float16* w_s1l1  = (_Float16*)(ws + off); off += 128*96*2;
  _Float16* w_s1l2  = (_Float16*)(ws + off); off += 128*128*2;
  _Float16* w_pr1   = (_Float16*)(ws + off); off += 128*416*2;
  _Float16* w_pr2   = (_Float16*)(ws + off); off += 128*128*2;

  auto conv = [&](const float* src, _Float16* dst, int srK, int cols, int Kpad){
    int total = cols*Kpad;
    convert_T_f16<<<dim3((total+255)/256), dim3(256), 0, stream>>>(src, dst, srK, cols, Kpad);
  };
  conv(stem2.W, w_stem2,  64,  64,  64);
  conv(s0l1.W,  w_s0l1,   91, 128,  96);
  conv(s0l2.W,  w_s0l2,  128, 128, 128);
  conv(s1l1.W,  w_s1l1,   91, 128,  96);
  conv(s1l2.W,  w_s1l2,  128, 128, 128);
  conv(pr1.W,   w_pr1,   416, 128, 416);
  conv(pr2.W,   w_pr2,   128, 128, 128);

  stem_kernel<<<dim3(cB*cN/128), dim3(256), 0, stream>>>(
      pc, stem1.W, stem1.b, stem1.g, stem1.be,
      stem2.b, stem2.g, stem2.be, w_stem2, feat);

  fps_kernel<<<dim3(cB), dim3(1024), 0, stream>>>(pc, cidx, pcen);

  ball_kernel<<<dim3(cB*cS/8), dim3(256), 0, stream>>>(pc, pcen, nidx);

  GroupArgs ga;
  ga.pc = pc; ga.feat = feat; ga.pcen = pcen; ga.nidx = nidx;
  ga.w0l1 = w_s0l1; ga.w0l2 = w_s0l2; ga.w1l1 = w_s1l1; ga.w1l2 = w_s1l2;
  ga.b0l1 = s0l1.b; ga.g0l1 = s0l1.g; ga.be0l1 = s0l1.be;
  ga.b0l2 = s0l2.b; ga.g0l2 = s0l2.g; ga.be0l2 = s0l2.be;
  ga.b1l1 = s1l1.b; ga.g1l1 = s1l1.g; ga.be1l1 = s1l1.be;
  ga.b1l2 = s1l2.b; ga.g1l2 = s1l2.g; ga.be1l2 = s1l2.be;
  ga.sfeat = sfeat;
  group_kernel<<<dim3(cB*cS), dim3(256), 0, stream>>>(ga);

  FinalArgs fa;
  fa.feat = feat; fa.cidx = cidx; fa.pcen = pcen; fa.sfeat = sfeat;
  fa.wp1 = w_pr1; fa.wp2 = w_pr2;
  fa.bp1 = pr1.b; fa.gp1 = pr1.g; fa.bep1 = pr1.be;
  fa.bp2 = pr2.b; fa.gp2 = pr2.g; fa.bep2 = pr2.be;
  fa.out = (float*)d_out;
  final_kernel<<<dim3(cB*cS/16), dim3(256), 0, stream>>>(fa);
}
